// VQEMA_25993142075435
// MI455X (gfx1250) — matrix operand order repaired and checked by reading, not compiled, not run on hardware
//
#include <hip/hip_runtime.h>

#define K_CODES 512
#define DIM     64
#define HW      4096            // 64*64
#define N_ROWS  131072          // 32*64*64
#define ROWS_PB 128
#define NBLK1   (N_ROWS / ROWS_PB)   // 1024
#define PADX    68              // padded LDS row stride (floats), 16B aligned
#define BETA_   0.25f
#define GAMMA_  0.99f
#define EPS_    1e-5f

typedef __attribute__((ext_vector_type(16))) _Float16 v16h;
typedef __attribute__((ext_vector_type(8)))  float    v8f;
typedef __attribute__((ext_vector_type(4)))  float    v4f;

// ---------------------------------------------------------------------------
// Kernel 1: nearest-codeword assignment via f16 WMMA, quantized output (in
// [B,D,H,W] layout), per-block commitment-loss partial, idx to ws + d_out.
// x tile is staged with gfx1250 async global->LDS copies (per-lane LDS dest
// addresses do the transpose + bank-padding scatter in the DMA itself).
// ---------------------------------------------------------------------------
__global__ __launch_bounds__(256)
void vq_assign_kernel(const float* __restrict__ enc,    // [B, D, H, W]
                      const float* __restrict__ E,      // [K, D]
                      float* __restrict__ out_q,        // [B, D, H, W]
                      float* __restrict__ out_idx_f,    // [N] (as float)
                      int*   __restrict__ ws_idx,       // [N]
                      float* __restrict__ ws_loss)      // [NBLK1]
{
    __shared__ float    s_x[ROWS_PB * PADX];   // x tile, f32, row-major [r][d]
    __shared__ _Float16 s_e[128 * DIM];        // codebook pass tile, f16
    __shared__ float    s_bias[128];           // 0.5*||e||^2 for pass tile
    __shared__ int      s_idx[ROWS_PB];
    __shared__ float    s_red[256];

    const int tid  = threadIdx.x;
    const int lane = tid & 31;
    const int wave = tid >> 5;
    const int half = lane >> 4;        // 0: lanes 0-15, 1: lanes 16-31
    const int l16  = lane & 15;
    const int n0   = blockIdx.x * ROWS_PB;     // first global row of block
    const int bimg = n0 >> 12;                 // n0 / HW (block never spans b)
    const int pb   = n0 & (HW - 1);            // pixel base within image

    const float* xb = enc + (size_t)bimg * DIM * HW + pb;

    // ---- Stage x tile via async DMA: enc[b, d, pb+r] -> s_x[r*PADX + d] ----
    // Global side is coalesced (consecutive lanes -> consecutive r); LDS side
    // is a per-lane scatter (transpose + padding) carried by the async unit.
    {
        const uint32_t lds_base = (uint32_t)(uintptr_t)(void*)&s_x[0];
        for (int i = tid; i < ROWS_PB * DIM; i += 256) {
            int r = i & (ROWS_PB - 1);
            int d = i >> 7;
            unsigned long long ga =
                (unsigned long long)(uintptr_t)(xb + (size_t)d * HW + r);
            uint32_t la = lds_base + (uint32_t)(r * PADX + d) * 4u;
            asm volatile("global_load_async_to_lds_b32 %0, %1, off"
                         :: "v"(la), "v"(ga)
                         : "memory");
        }
        asm volatile("s_wait_asynccnt 0" ::: "memory");
    }
    __syncthreads();

    // ---- Build A fragments (16 rows x 32 K, f16) once; reused all passes --
    // Lanes 0-15: V0-7 = K[c*32+0..15]
    // Lanes16-31: V0-7 = K[c*32+16..31]
    v16h afrag[2];
    {
        const int rl = wave * 16 + l16;            // local row in tile
        const float* xr = &s_x[rl * PADX];
        #pragma unroll
        for (int c = 0; c < 2; ++c) {
            const float* p0 = xr + c * 32 + half * 16;
            const float* p1 = xr + c * 32 + half * 16 + 8;
            v4f f0 = *(const v4f*)(p0);
            v4f f1 = *(const v4f*)(p0 + 4);
            v4f f2 = *(const v4f*)(p1);
            v4f f3 = *(const v4f*)(p1 + 4);
            #pragma unroll
            for (int j = 0; j < 4; ++j) {
                afrag[c][j]      = (_Float16)f0[j];
                afrag[c][4 + j]  = (_Float16)f1[j];
                afrag[c][8 + j]  = (_Float16)f2[j];
                afrag[c][12 + j] = (_Float16)f3[j];
            }
        }
    }

    float bestv[8];
    int   besti[8];
    #pragma unroll
    for (int s = 0; s < 8; ++s) { bestv[s] = -3.4e38f; besti[s] = 0; }

    // ---- Sweep codebook: 4 passes x 128 codewords in LDS ----
    #pragma unroll 1
    for (int p = 0; p < 4; ++p) {
        __syncthreads();
        for (int i = tid; i < 128 * DIM; i += 256)
            s_e[i] = (_Float16)E[p * 128 * DIM + i];
        if (p < 3)   // prefetch next codebook slab
            __builtin_prefetch(E + (p + 1) * 128 * DIM + tid * 32, 0, 0);
        __syncthreads();
        if (tid < 128) {
            float s = 0.f;
            #pragma unroll
            for (int d = 0; d < DIM; ++d) {
                float v = (float)s_e[tid * DIM + d];
                s += v * v;
            }
            s_bias[tid] = 0.5f * s;
        }
        __syncthreads();

        // unroll 2: two independent acc chains cover the WMMA->VALU hazard
        #pragma unroll 2
        for (int kt = 0; kt < 8; ++kt) {
            const int col = kt * 16 + l16;         // codeword within pass
            const float bv = s_bias[col];
            const _Float16* eb = &s_e[col * DIM];
            // B fragments 32x16: lanes0-15 K=0..15 of chunk, lanes16-31 K=16..31
            v16h b0 = *(const v16h*)(eb + half * 16);
            v16h b1 = *(const v16h*)(eb + 32 + half * 16);
            v8f acc;
            #pragma unroll
            for (int s = 0; s < 8; ++s) acc[s] = -bv;  // C = -0.5*||e||^2
            acc = __builtin_amdgcn_wmma_f32_16x16x32_f16(
                      false, afrag[0], false, b0, (short)0, acc, false, false);
            acc = __builtin_amdgcn_wmma_f32_16x16x32_f16(
                      false, afrag[1], false, b1, (short)0, acc, false, false);
            const int kg = p * 128 + col;
            #pragma unroll
            for (int s = 0; s < 8; ++s) {
                bool better = acc[s] > bestv[s];   // scan ascending: ties keep low k
                bestv[s] = better ? acc[s] : bestv[s];
                besti[s] = better ? kg     : besti[s];
            }
        }
    }

    // ---- Cross-lane argmax within each 16-lane group (smaller k on ties) --
    #pragma unroll
    for (int s = 0; s < 8; ++s) {
        float bv = bestv[s];
        int   bi = besti[s];
        #pragma unroll
        for (int off = 1; off < 16; off <<= 1) {
            float ov = __shfl_xor(bv, off, 32);
            int   oi = __shfl_xor(bi, off, 32);
            if (ov > bv || (ov == bv && oi < bi)) { bv = ov; bi = oi; }
        }
        bestv[s] = bv; besti[s] = bi;
    }
    if (l16 == 0) {
        #pragma unroll
        for (int s = 0; s < 8; ++s)
            s_idx[wave * 16 + half * 8 + s] = besti[s];  // M = s (+8 for hi half)
    }
    __syncthreads();

    // ---- Output phase: quantized (transposed, coalesced) + loss partial ---
    float* outb = out_q + (size_t)bimg * DIM * HW + pb;
    float lsum = 0.f;
    for (int i = tid; i < ROWS_PB * DIM; i += 256) {
        int r = i & (ROWS_PB - 1);
        int d = i >> 7;
        int k = s_idx[r];
        float q  = E[k * DIM + d];
        float xv = s_x[r * PADX + d];
        float df = q - xv;
        lsum += df * df;
        outb[(size_t)d * HW + r] = q;
    }
    if (tid < ROWS_PB) {
        ws_idx[n0 + tid]    = s_idx[tid];
        out_idx_f[n0 + tid] = (float)s_idx[tid];
    }
    s_red[tid] = lsum;
    __syncthreads();
    for (int s = 128; s > 0; s >>= 1) {
        if (tid < s) s_red[tid] += s_red[tid + s];
        __syncthreads();
    }
    if (tid == 0) ws_loss[blockIdx.x] = s_red[0];
}

// ---------------------------------------------------------------------------
// Kernel 2: deterministic segment sums (one wave per codeword) + EMA for m,M.
// ---------------------------------------------------------------------------
__global__ __launch_bounds__(256)
void vq_ema_kernel(const float* __restrict__ enc,      // [B, D, H, W]
                   const int*   __restrict__ ws_idx,   // [N]
                   const float* __restrict__ ema_m,    // [K, D]
                   const float* __restrict__ ema_M,    // [K]
                   float* __restrict__ out_new_m,      // [K, D]
                   float* __restrict__ ws_Mpre)        // [K]
{
    const int lane = threadIdx.x & 31;
    const int wave = threadIdx.x >> 5;
    const int k    = blockIdx.x * 8 + wave;            // 64 blocks * 8 waves = 512

    float acc0 = 0.f, acc1 = 0.f, cnt = 0.f;
    for (int base = 0; base < N_ROWS; base += 32) {
        int v = ws_idx[base + lane];
        unsigned long long mask = __ballot(v == k);
        while (mask) {                                  // ascending row order
            int l = __ffsll((unsigned long long)mask) - 1;
            mask &= mask - 1;
            int row = base + l;
            int bb  = row >> 12;                        // row / HW
            int pp  = row & (HW - 1);
            const float* xr = enc + (size_t)bb * DIM * HW + pp;
            acc0 += xr[(size_t)lane * HW];
            acc1 += xr[(size_t)(lane + 32) * HW];
            cnt  += 1.f;
        }
    }
    out_new_m[k * DIM + lane]      = GAMMA_ * ema_m[k * DIM + lane]      + (1.f - GAMMA_) * acc0;
    out_new_m[k * DIM + 32 + lane] = GAMMA_ * ema_m[k * DIM + 32 + lane] + (1.f - GAMMA_) * acc1;
    if (lane == 0)
        ws_Mpre[k] = GAMMA_ * ema_M[k] + (1.f - GAMMA_) * cnt;
}

// ---------------------------------------------------------------------------
// Kernel 3: loss finalize, Laplace-smoothed new_M, new_embeddings = m / M.
// ---------------------------------------------------------------------------
__global__ __launch_bounds__(512)
void vq_final_kernel(const float* __restrict__ ws_loss,   // [NBLK1]
                     const float* __restrict__ ws_Mpre,   // [K]
                     const float* __restrict__ new_m,     // [K, D] (d_out)
                     float* __restrict__ out_loss,        // [1]
                     float* __restrict__ out_emb,         // [K, D]
                     float* __restrict__ out_M)           // [K]
{
    __shared__ float s1[512];
    __shared__ float s_ntot;
    const int tid = threadIdx.x;

    // loss = BETA * sum(partials) / (N*D)
    s1[tid] = ws_loss[tid] + ws_loss[tid + 512];
    __syncthreads();
    for (int s = 256; s > 0; s >>= 1) {
        if (tid < s) s1[tid] += s1[tid + s];
        __syncthreads();
    }
    if (tid == 0)
        out_loss[0] = BETA_ * s1[0] * (1.0f / (float)((size_t)N_ROWS * DIM));
    __syncthreads();

    // N_tot = sum(M_pre)
    s1[tid] = ws_Mpre[tid];
    __syncthreads();
    for (int s = 256; s > 0; s >>= 1) {
        if (tid < s) s1[tid] += s1[tid + s];
        __syncthreads();
    }
    if (tid == 0) s_ntot = s1[0];
    __syncthreads();

    const float ntot  = s_ntot;
    const float denom = ntot + EPS_ * (float)K_CODES;

    {   // new_M
        float nM = (ws_Mpre[tid] + EPS_) / denom * ntot;
        out_M[tid] = nM;
    }
    for (int i = tid; i < K_CODES * DIM; i += 512) {
        int k = i >> 6;
        float nM = (ws_Mpre[k] + EPS_) / denom * ntot;
        out_emb[i] = new_m[i] / nM;
    }
}

// ---------------------------------------------------------------------------
extern "C" void kernel_launch(void* const* d_in, const int* in_sizes, int n_in,
                              void* d_out, int out_size, void* d_ws, size_t ws_size,
                              hipStream_t stream)
{
    (void)in_sizes; (void)n_in; (void)out_size; (void)ws_size;

    const float* enc   = (const float*)d_in[0];   // [32, 64, 64, 64]
    const float* emb   = (const float*)d_in[1];   // [512, 64]
    const float* ema_m = (const float*)d_in[2];   // [512, 64]
    const float* ema_M = (const float*)d_in[3];   // [512, 1]

    // d_out: out | loss | idx | new_embeddings | new_m | new_M (flat, f32)
    float* out_q    = (float*)d_out;
    float* out_loss = out_q + (size_t)32 * DIM * HW;   // 8388608
    float* out_idx  = out_loss + 1;
    float* out_emb  = out_idx + N_ROWS;
    float* out_m    = out_emb + K_CODES * DIM;
    float* out_M    = out_m + K_CODES * DIM;

    int*   ws_idx  = (int*)d_ws;                       // N ints
    float* ws_loss = (float*)d_ws + N_ROWS;            // NBLK1 floats
    float* ws_Mpre = ws_loss + NBLK1;                  // K floats

    vq_assign_kernel<<<NBLK1, 256, 0, stream>>>(enc, emb, out_q, out_idx,
                                                ws_idx, ws_loss);
    vq_ema_kernel<<<K_CODES / 8, 256, 0, stream>>>(enc, ws_idx, ema_m, ema_M,
                                                   out_m, ws_Mpre);
    vq_final_kernel<<<1, 512, 0, stream>>>(ws_loss, ws_Mpre, out_m,
                                           out_loss, out_emb, out_M);
}
